// SimpleTransformer_attention_62706522521861
// MI455X (gfx1250) — compile-verified, matrix operand order checked by reading, not run on hardware
//
#include <hip/hip_runtime.h>
#include <stdint.h>

#define B_  32
#define N_  4096
#define H_  8
#define Dh_ 64
#define D_  512
#define MT  64      // rows per workgroup tile in projection GEMM
#define LDSW 520    // padded LDS row stride (bf16 elems): +8 -> 4-bank rotation/row

typedef __attribute__((ext_vector_type(16))) __bf16 v16bf;
typedef __attribute__((ext_vector_type(8)))  float  v8f;

struct __align__(16) U128 { uint32_t x, y, z, w; };
struct __align__(16) F128 { float x, y, z, w; };
union Frag { U128 u[2]; v16bf v; };

__device__ __forceinline__ uint16_t f32_bf16(float x) {
    uint32_t u = __float_as_uint(x);
    uint32_t r = u + 0x7FFFu + ((u >> 16) & 1u);   // round-to-nearest-even
    return (uint16_t)(r >> 16);
}
__device__ __forceinline__ float bf16_f32(uint16_t h) {
    return __uint_as_float(((uint32_t)h) << 16);
}

// ---------------------------------------------------------------------------
// Pack combined per-head weights into wave32 WMMA B-fragment tiles.
// Tile (ct,kt): 16 cols x 32 K, 1024B. Lane l owns col n=ct*16+(l&15);
// dwords [g*4+j] hold bf16 pair (K = kt*32 + g*16 + 8*(l>>4) + 2j , K+1).
// Column c<64 comes from the 512x64 "self" weight, else from the 512x512.
// ---------------------------------------------------------------------------
__global__ void pack_w(const float* __restrict__ Wc, const float* __restrict__ Ws,
                       uint32_t* __restrict__ out, int ncols) {
    int idx = blockIdx.x * 256 + threadIdx.x;
    if (idx >= ncols * 256) return;
    int j    = idx & 3;
    int g    = (idx >> 2) & 1;
    int lane = (idx >> 3) & 31;
    int kt   = (idx >> 8) & 15;
    int ct   = idx >> 12;
    int k = kt * 32 + g * 16 + 8 * (lane >> 4) + 2 * j;
    int c = ct * 16 + (lane & 15);
    float w0, w1;
    if (c < 64) { w0 = Ws[k * 64 + c];  w1 = Ws[(k + 1) * 64 + c]; }
    else        { w0 = Wc[k * 512 + c]; w1 = Wc[(k + 1) * 512 + c]; }
    out[idx] = (uint32_t)f32_bf16(w0) | ((uint32_t)f32_bf16(w1) << 16);
}

// ---------------------------------------------------------------------------
// Fused projection GEMM: per 64-row tile,
//   phase0 (A = key+pos):  K all heads (Wk_s|Wk_c), V head0 (Wv_s), Q-self sum (Wq_s)
//   phase1 (A = value+pos): V heads 1..7 (Wv_c)
// bf16 WMMA 16x16x32; each wave owns a 16-col strip across 4 M-subtiles so the
// B fragment is reused by 4 WMMAs. Stores bf16 K/V in [b][h][n][dh].
// ---------------------------------------------------------------------------
__global__ __launch_bounds__(256)
void proj_kernel(const float* __restrict__ key, const float* __restrict__ value,
                 const float* __restrict__ pos,
                 const uint32_t* __restrict__ WKp, const uint32_t* __restrict__ WVp,
                 const uint32_t* __restrict__ WQp,
                 const float* __restrict__ bk_c, const float* __restrict__ bk_s,
                 const float* __restrict__ bv_c, const float* __restrict__ bv_s,
                 uint16_t* __restrict__ Kbuf, uint16_t* __restrict__ Vbuf,
                 float* __restrict__ q_accum) {
    __shared__ __align__(16) uint16_t At[MT * LDSW];   // 66,560 B
    const int tid  = threadIdx.x;
    const int wave = tid >> 5, lane = tid & 31;
    const int ln   = lane & 15, khi = lane >> 4;
    const int r0   = blockIdx.x * MT;      // global row base (flattened b*N+n)
    const int b    = r0 >> 12;             // N=4096, tile never crosses b

    for (int p = 0; p < 2; ++p) {
        const float* src = p ? value : key;
        // vectorized A-tile staging: float4 loads, packed bf16-pair LDS stores
        for (int i = tid; i < MT * (D_ / 4); i += 256) {
            int row = i >> 7;              // 128 float4 per row
            int c4  = (i & 127) << 2;
            int r = r0 + row, n = r & (N_ - 1);
            F128 a = *(const F128*)(src + (size_t)r * D_ + c4);
            F128 q = *(const F128*)(pos + (size_t)n * D_ + c4);
            uint32_t lo = (uint32_t)f32_bf16(a.x + q.x) | ((uint32_t)f32_bf16(a.y + q.y) << 16);
            uint32_t hi = (uint32_t)f32_bf16(a.z + q.z) | ((uint32_t)f32_bf16(a.w + q.w) << 16);
            uint32_t* d = (uint32_t*)&At[row * LDSW + c4];
            d[0] = lo; d[1] = hi;
        }
        __syncthreads();

        const int nstrips = p ? 28 : 40;
        for (int js = wave; js < nstrips; js += 8) {
            int kind, ct;                    // 0=K, 1=V, 2=Q-self
            const uint32_t* Wp;
            if (p == 1)       { kind = 1; ct = js + 4;  Wp = WVp; }
            else if (js < 32) { kind = 0; ct = js;      Wp = WKp; }
            else if (js < 36) { kind = 1; ct = js - 32; Wp = WVp; }
            else              { kind = 2; ct = js - 36; Wp = WQp; }

            v8f acc[4];
            #pragma unroll
            for (int st = 0; st < 4; ++st)
                #pragma unroll
                for (int v = 0; v < 8; ++v) acc[st][v] = 0.f;

            const uint32_t* wbase = Wp + ((size_t)(ct * 16 * 32 + lane) << 3);
            #pragma unroll 4
            for (int kt = 0; kt < 16; ++kt) {
                Frag bf;
                bf.u[0] = *(const U128*)(wbase + (kt << 8));
                bf.u[1] = *(const U128*)(wbase + (kt << 8) + 4);
                const int ab = kt * 32 + khi * 8;
                #pragma unroll
                for (int st = 0; st < 4; ++st) {
                    Frag a;
                    const int ro = (st * 16 + ln) * LDSW + ab;
                    a.u[0] = *(const U128*)&At[ro];
                    a.u[1] = *(const U128*)&At[ro + 16];
                    acc[st] = __builtin_amdgcn_wmma_f32_16x16x32_bf16(
                                  false, a.v, false, bf.v, (short)0, acc[st], false, false);
                }
            }

            if (kind == 2) {                       // Q-self: sum over rows -> mean later
                float s = 0.f;
                #pragma unroll
                for (int st = 0; st < 4; ++st)
                    #pragma unroll
                    for (int v = 0; v < 8; ++v) s += acc[st][v];
                atomicAdd(&q_accum[b * 64 + ct * 16 + ln], s);
            } else {
                const bool self = (ct < 4);        // wave-uniform: strip entirely head0
                int c = ct * 16 + ln;
                int h = c >> 6, dh = c & 63;
                const float* bsrc = (kind == 0) ? (self ? bk_s : bk_c)
                                                : (self ? bv_s : bv_c);
                float bias = bsrc[c];
                uint16_t* Out = (kind == 0) ? Kbuf : Vbuf;
                size_t base = (((size_t)b * H_ + h) * N_) * Dh_ + dh;
                #pragma unroll
                for (int st = 0; st < 4; ++st) {
                    #pragma unroll
                    for (int v = 0; v < 8; ++v) {
                        int m = st * 16 + v + 8 * khi;
                        int n = (r0 + m) & (N_ - 1);
                        Out[base + (size_t)n * Dh_] = f32_bf16(acc[st][v] + bias);
                    }
                }
            }
        }
        __syncthreads();
    }
}

// ---------------------------------------------------------------------------
// q heads: head0 = mean(q_proj_self) = accum/N + bq_s ; heads1..7 = query@Wq_c.
// ---------------------------------------------------------------------------
__global__ __launch_bounds__(512)
void qfinal_kernel(const float* __restrict__ query,
                   const float* __restrict__ Wq_c, const float* __restrict__ bq_c,
                   const float* __restrict__ bq_s,
                   const float* __restrict__ q_accum,
                   float* __restrict__ q_buf) {
    __shared__ float qs[D_];
    int b = blockIdx.x, c = threadIdx.x;
    qs[c] = query[b * D_ + c];
    __syncthreads();
    float r;
    if (c < 64) {
        r = q_accum[b * 64 + c] * (1.0f / N_) + bq_s[c];
    } else {
        float s = bq_c[c];
        for (int k = 0; k < D_; ++k) s = fmaf(qs[k], Wq_c[k * D_ + c], s);
        r = s;
    }
    q_buf[b * D_ + c] = r;
}

// ---------------------------------------------------------------------------
// Attention: one block per (b,h). scores -> softmax -> weights (to d_out) -> out.
// ---------------------------------------------------------------------------
__global__ __launch_bounds__(256)
void attn_kernel(const float* __restrict__ q_buf,
                 const uint16_t* __restrict__ Kbuf, const uint16_t* __restrict__ Vbuf,
                 float* __restrict__ out) {
    __shared__ float sc[N_];
    __shared__ float red[256];
    __shared__ float qs[Dh_];
    const int bh = blockIdx.x;             // b*H + h
    const int tid = threadIdx.x;
    if (tid < Dh_) qs[tid] = q_buf[bh * Dh_ + tid];
    __syncthreads();

    float lmax = -3.4e38f;
    for (int n = tid; n < N_; n += 256) {
        const uint32_t* kr = (const uint32_t*)(Kbuf + ((size_t)bh * N_ + n) * Dh_);
        float s = 0.f;
        #pragma unroll
        for (int j = 0; j < 32; ++j) {
            uint32_t u = kr[j];
            s = fmaf(bf16_f32((uint16_t)u),         qs[2 * j],     s);
            s = fmaf(bf16_f32((uint16_t)(u >> 16)), qs[2 * j + 1], s);
        }
        s *= 0.125f;                        // 1/sqrt(Dh), TEMPERATURE=1
        sc[n] = s;
        lmax = fmaxf(lmax, s);
    }
    red[tid] = lmax; __syncthreads();
    for (int st = 128; st > 0; st >>= 1) {
        if (tid < st) red[tid] = fmaxf(red[tid], red[tid + st]);
        __syncthreads();
    }
    float gmax = red[0];
    __syncthreads();

    float lsum = 0.f;
    for (int n = tid; n < N_; n += 256) {
        float e = __expf(sc[n] - gmax);
        sc[n] = e; lsum += e;
    }
    red[tid] = lsum; __syncthreads();
    for (int st = 128; st > 0; st >>= 1) {
        if (tid < st) red[tid] += red[tid + st];
        __syncthreads();
    }
    float inv = 1.0f / red[0];
    __syncthreads();

    float* wout = out + B_ * D_ + (size_t)bh * N_;
    for (int n = tid; n < N_; n += 256) {
        float w = sc[n] * inv;
        sc[n] = w;
        wout[n] = w;
    }
    __syncthreads();

    const int dh = tid & 63, chunk = tid >> 6;
    float acc = 0.f;
    const uint16_t* vb = Vbuf + (size_t)bh * N_ * Dh_ + dh;
    for (int n = chunk * 1024; n < chunk * 1024 + 1024; ++n)
        acc = fmaf(sc[n], bf16_f32(vb[(size_t)n * Dh_]), acc);
    red[tid] = acc; __syncthreads();
    if (tid < 64) {
        out[bh * Dh_ + tid] = red[tid] + red[tid + 64] + red[tid + 128] + red[tid + 192];
    }
}

extern "C" void kernel_launch(void* const* d_in, const int* in_sizes, int n_in,
                              void* d_out, int out_size, void* d_ws, size_t ws_size,
                              hipStream_t stream) {
    (void)in_sizes; (void)n_in; (void)out_size; (void)ws_size;
    const float* query = (const float*)d_in[0];
    const float* key   = (const float*)d_in[1];
    const float* value = (const float*)d_in[2];
    const float* pos   = (const float*)d_in[3];
    const float* Wq_c  = (const float*)d_in[4];
    const float* bq_c  = (const float*)d_in[5];
    const float* Wk_c  = (const float*)d_in[6];
    const float* bk_c  = (const float*)d_in[7];
    const float* Wv_c  = (const float*)d_in[8];
    const float* bv_c  = (const float*)d_in[9];
    const float* Wq_s  = (const float*)d_in[10];
    const float* bq_s  = (const float*)d_in[11];
    const float* Wk_s  = (const float*)d_in[12];
    const float* bk_s  = (const float*)d_in[13];
    const float* Wv_s  = (const float*)d_in[14];
    const float* bv_s  = (const float*)d_in[15];
    float* out = (float*)d_out;

    char* ws = (char*)d_ws;
    size_t off = 0;
    auto carve = [&](size_t bytes) -> void* {
        void* p = ws + off;
        off += (bytes + 255) & ~(size_t)255;
        return p;
    };
    uint16_t* Kbuf  = (uint16_t*)carve((size_t)B_ * H_ * N_ * Dh_ * 2);  // 128 MB
    uint16_t* Vbuf  = (uint16_t*)carve((size_t)B_ * H_ * N_ * Dh_ * 2);  // 128 MB
    uint32_t* WKp   = (uint32_t*)carve((size_t)512 * 256 * 4);
    uint32_t* WVp   = (uint32_t*)carve((size_t)512 * 256 * 4);
    uint32_t* WQp   = (uint32_t*)carve((size_t)64 * 256 * 4);
    float* q_accum  = (float*)carve((size_t)B_ * 64 * 4);
    float* q_buf    = (float*)carve((size_t)B_ * D_ * 4);

    hipMemsetAsync(q_accum, 0, (size_t)B_ * 64 * 4, stream);
    pack_w<<<512, 256, 0, stream>>>(Wk_c, Wk_s, WKp, 512);
    pack_w<<<512, 256, 0, stream>>>(Wv_c, Wv_s, WVp, 512);
    pack_w<<<64, 256, 0, stream>>>(Wq_s, Wq_s, WQp, 64);
    proj_kernel<<<(B_ * N_) / MT, 256, 0, stream>>>(key, value, pos, WKp, WVp, WQp,
                                                    bk_c, bk_s, bv_c, bv_s,
                                                    Kbuf, Vbuf, q_accum);
    qfinal_kernel<<<B_, 512, 0, stream>>>(query, Wq_c, bq_c, bq_s, q_accum, q_buf);
    attn_kernel<<<B_ * H_, 256, 0, stream>>>(q_buf, Kbuf, Vbuf, out);
}
